// Interest_45414984188488
// MI455X (gfx1250) — matrix operand order repaired and checked by reading, not compiled, not run on hardware
//
#include <hip/hip_runtime.h>
#include <hip/hip_bf16.h>

// ---------------------------------------------------------------------------
// DIEN (GRU -> DIN attention -> AUGRU) for MI455X / gfx1250, wave32 + WMMA.
// All GEMMs: v_wmma_f32_16x16x32_bf16 (bf16 in, f32 accumulate).
// Scans: one block = 16 batch rows, weights resident in VGPRs (1024-VGPR mode),
// x-tiles software-pipelined (prefetch t+1 during step t's WMMAs).
// interests kept bf16-only (52 MB -> lives in 192 MB L2 for re-reads).
// ---------------------------------------------------------------------------

typedef __attribute__((ext_vector_type(16))) __bf16 v16bf;
typedef __attribute__((ext_vector_type(8)))  float  v8f;

#define B_  1024
#define T_  200
#define H_  128
#define G3H 384          // 3*H gates (r,z,n)

constexpr int GRU_PACK = 24 * 4 * 32 * 16;  // 24 n-tiles x 4 k-blocks (K=128)
constexpr int W1_PACK  = 6 * 16 * 32 * 16;  // 6 n-tiles (80 pad 96) x 16 k-blocks (K=512)

__device__ __forceinline__ float sigmoidf_(float x) {
    return 1.0f / (1.0f + __expf(-x));
}

// A-fragment K index, contiguous order k = j + 16*hlf (same order as the packed B)
__device__ __forceinline__ int a_klocal(int j, int hlf) {
    int v = j >> 1, p = j & 1;
    return 2 * v + p + 16 * hlf;
}

// ---------------------------------------------------------------------------
// Pack f32 weights -> bf16 B-fragments: [n_tile][k_block][lane(32)][16 bf16]
// B order used here (32x16): lane = half*16 + n%16 ; K = 2v+p + 16*half (contiguous)
// ---------------------------------------------------------------------------
__global__ __launch_bounds__(256) void pack_weights_kernel(
    const float* __restrict__ Wih_e, const float* __restrict__ Whh_e,
    const float* __restrict__ Wih_a, const float* __restrict__ Whh_a,
    const float* __restrict__ W1,
    __bf16* __restrict__ pWih_e, __bf16* __restrict__ pWhh_e,
    __bf16* __restrict__ pWih_a, __bf16* __restrict__ pWhh_a,
    __bf16* __restrict__ pW1)
{
    int gid = blockIdx.x * blockDim.x + threadIdx.x;
    if (gid < 4 * GRU_PACK) {
        int sel = gid / GRU_PACK;
        int idx = gid % GRU_PACK;
        int j = idx & 15, lane = (idx >> 4) & 31;
        int kb = (idx >> 9) & 3, nt = idx >> 11;
        int n   = nt * 16 + (lane & 15);
        int hlf = lane >> 4;
        int v = j >> 1, p = j & 1;
        int k = kb * 32 + 2 * v + p + 16 * hlf;
        const float* src = (sel == 0) ? Wih_e : (sel == 1) ? Whh_e
                         : (sel == 2) ? Wih_a : Whh_a;
        __bf16* dst = (sel == 0) ? pWih_e : (sel == 1) ? pWhh_e
                    : (sel == 2) ? pWih_a : pWhh_a;
        dst[idx] = (__bf16)src[n * H_ + k];
    } else if (gid < 4 * GRU_PACK + W1_PACK) {
        int idx = gid - 4 * GRU_PACK;
        int j = idx & 15, lane = (idx >> 4) & 31;
        int kb = (idx >> 9) & 15, nt = idx >> 13;
        int n   = nt * 16 + (lane & 15);
        int hlf = lane >> 4;
        int v = j >> 1, p = j & 1;
        int k = kb * 32 + 2 * v + p + 16 * hlf;
        pW1[idx] = (__bf16)((n < 80) ? W1[n * 512 + k] : 0.0f);
    }
}

// ---------------------------------------------------------------------------
// GRU scan. Thread tid owns column k=tid of the 16-row tile; h (f32) lives in
// 16 registers per thread; bf16 h copy + x tile in LDS feed WMMA A-fragments.
// ---------------------------------------------------------------------------
__global__ __launch_bounds__(128) void gru_scan_kernel(
    const float* __restrict__ keys, const int* __restrict__ keys_len,
    const float* __restrict__ bih, const float* __restrict__ bhh,
    const __bf16* __restrict__ pWih, const __bf16* __restrict__ pWhh,
    __bf16* __restrict__ interests_bf)
{
    __shared__ __bf16 xt[16][H_];
    __shared__ __bf16 hb[16][H_];
    __shared__ float  si[16][G3H];
    __shared__ float  sh[16][G3H];
    __shared__ int    lens[16];

    const int tid  = threadIdx.x;
    const int lane = tid & 31;
    const int wv   = tid >> 5;            // 0..3
    const int b0   = blockIdx.x * 16;
    const int mrow = lane & 15;
    const int hlf  = lane >> 4;

    // resident B-fragments (per wave: 6 n-tiles x 4 k-blocks x 2 matrices)
    v16bf wbi[6][4], wbh[6][4];
    #pragma unroll
    for (int nti = 0; nti < 6; ++nti) {
        int nt = wv * 6 + nti;
        #pragma unroll
        for (int kb = 0; kb < 4; ++kb) {
            wbi[nti][kb] = *(const v16bf*)&pWih[(size_t)(((nt * 4 + kb) * 32) + lane) * 16];
            wbh[nti][kb] = *(const v16bf*)&pWhh[(size_t)(((nt * 4 + kb) * 32) + lane) * 16];
        }
    }

    // hidden state column k=tid for all 16 rows, in registers
    float hreg[16];
    #pragma unroll
    for (int m = 0; m < 16; ++m) hreg[m] = 0.0f;
    for (int idx = tid; idx < 16 * H_; idx += 128) {
        int m = idx >> 7, k = idx & 127;
        hb[m][k] = (__bf16)0.0f;
    }
    if (tid < 16) lens[tid] = keys_len[b0 + tid];

    // hoist biases (column k=tid)
    const float bir = bih[tid],        bhr = bhh[tid];
    const float biz = bih[H_ + tid],   bhz = bhh[H_ + tid];
    const float bin_ = bih[2*H_ + tid], bhn = bhh[2*H_ + tid];

    // prefetch t=0 x tile into registers
    float xreg[16];
    #pragma unroll
    for (int m = 0; m < 16; ++m)
        xreg[m] = keys[((size_t)(b0 + m) * T_ + 0) * H_ + tid];
    __syncthreads();

    for (int t = 0; t < T_; ++t) {
        // commit prefetched x tile to LDS (bf16)
        #pragma unroll
        for (int m = 0; m < 16; ++m)
            xt[m][tid] = (__bf16)xreg[m];
        __syncthreads();

        // issue next step's loads now; wait lands at next iteration's commit
        if (t + 1 < T_) {
            #pragma unroll
            for (int m = 0; m < 16; ++m)
                xreg[m] = keys[((size_t)(b0 + m) * T_ + (t + 1)) * H_ + tid];
        }
        // L2 prefetch at distance 8 (covers the tile's cachelines)
        if (t + 8 < T_)
            __builtin_prefetch(&keys[((size_t)(b0 + (tid >> 3)) * T_ + (t + 8)) * H_
                                     + (tid & 7) * 16], 0, 1);

        // A fragments (shared across this wave's 6 n-tiles)
        v16bf ax[4], ah[4];
        #pragma unroll
        for (int kb = 0; kb < 4; ++kb) {
            #pragma unroll
            for (int j = 0; j < 16; ++j) {
                int k = kb * 32 + a_klocal(j, hlf);
                ax[kb][j] = xt[mrow][k];
                ah[kb][j] = hb[mrow][k];
            }
        }

        #pragma unroll
        for (int nti = 0; nti < 6; ++nti) {
            int nt = wv * 6 + nti;
            v8f accI = {}; v8f accH = {};
            #pragma unroll
            for (int kb = 0; kb < 4; ++kb) {
                accI = __builtin_amdgcn_wmma_f32_16x16x32_bf16(
                           false, ax[kb], false, wbi[nti][kb], (short)0, accI, false, false);
                accH = __builtin_amdgcn_wmma_f32_16x16x32_bf16(
                           false, ah[kb], false, wbh[nti][kb], (short)0, accH, false, false);
            }
            int n = nt * 16 + (lane & 15);
            #pragma unroll
            for (int v = 0; v < 8; ++v) {
                int m = v + 8 * hlf;
                si[m][n] = accI[v];
                sh[m][n] = accH[v];
            }
        }
        __syncthreads();

        // gate math: thread tid handles column k=tid, rows m=0..15
        #pragma unroll
        for (int m = 0; m < 16; ++m) {
            float r  = sigmoidf_((si[m][tid]          + bir) + (sh[m][tid]          + bhr));
            float z  = sigmoidf_((si[m][H_ + tid]     + biz) + (sh[m][H_ + tid]     + bhz));
            float ghn =           sh[m][2 * H_ + tid] + bhn;
            float ng = tanhf((si[m][2 * H_ + tid] + bin_) + r * ghn);
            float hprev = hreg[m];
            float hnew  = (1.0f - z) * ng + z * hprev;
            bool valid  = t < lens[m];
            float hkeep = valid ? hnew : hprev;
            hreg[m] = hkeep;
            hb[m][tid] = (__bf16)hkeep;
            interests_bf[((size_t)(b0 + m) * T_ + t) * H_ + tid] =
                (__bf16)(valid ? hnew : 0.0f);
        }
        __syncthreads();
    }
}

// ---------------------------------------------------------------------------
// AUGRU scan: same structure; u gate scaled by attention score, gates NEW state.
// Input x = interests (bf16). Writes final hidden state to out[B,H] (f32).
// ---------------------------------------------------------------------------
__global__ __launch_bounds__(128) void augru_scan_kernel(
    const __bf16* __restrict__ xin_bf, const float* __restrict__ att,
    const int* __restrict__ keys_len,
    const float* __restrict__ bih, const float* __restrict__ bhh,
    const __bf16* __restrict__ pWih, const __bf16* __restrict__ pWhh,
    float* __restrict__ out)
{
    __shared__ __bf16 xt[16][H_];
    __shared__ __bf16 hb[16][H_];
    __shared__ float  si[16][G3H];
    __shared__ float  sh[16][G3H];
    __shared__ int    lens[16];
    __shared__ float  att_s[16];

    const int tid  = threadIdx.x;
    const int lane = tid & 31;
    const int wv   = tid >> 5;
    const int b0   = blockIdx.x * 16;
    const int mrow = lane & 15;
    const int hlf  = lane >> 4;

    v16bf wbi[6][4], wbh[6][4];
    #pragma unroll
    for (int nti = 0; nti < 6; ++nti) {
        int nt = wv * 6 + nti;
        #pragma unroll
        for (int kb = 0; kb < 4; ++kb) {
            wbi[nti][kb] = *(const v16bf*)&pWih[(size_t)(((nt * 4 + kb) * 32) + lane) * 16];
            wbh[nti][kb] = *(const v16bf*)&pWhh[(size_t)(((nt * 4 + kb) * 32) + lane) * 16];
        }
    }

    float hreg[16];
    #pragma unroll
    for (int m = 0; m < 16; ++m) hreg[m] = 0.0f;
    for (int idx = tid; idx < 16 * H_; idx += 128) {
        int m = idx >> 7, k = idx & 127;
        hb[m][k] = (__bf16)0.0f;
    }
    if (tid < 16) lens[tid] = keys_len[b0 + tid];

    const float bir = bih[tid],        bhr = bhh[tid];
    const float biu = bih[H_ + tid],   bhu = bhh[H_ + tid];
    const float bin_ = bih[2*H_ + tid], bhn = bhh[2*H_ + tid];

    __bf16 xreg[16];
    #pragma unroll
    for (int m = 0; m < 16; ++m)
        xreg[m] = xin_bf[((size_t)(b0 + m) * T_ + 0) * H_ + tid];
    __syncthreads();

    for (int t = 0; t < T_; ++t) {
        #pragma unroll
        for (int m = 0; m < 16; ++m)
            xt[m][tid] = xreg[m];
        if (tid < 16) att_s[tid] = att[(size_t)(b0 + tid) * T_ + t];
        __syncthreads();

        if (t + 1 < T_) {
            #pragma unroll
            for (int m = 0; m < 16; ++m)
                xreg[m] = xin_bf[((size_t)(b0 + m) * T_ + (t + 1)) * H_ + tid];
        }

        v16bf ax[4], ah[4];
        #pragma unroll
        for (int kb = 0; kb < 4; ++kb) {
            #pragma unroll
            for (int j = 0; j < 16; ++j) {
                int k = kb * 32 + a_klocal(j, hlf);
                ax[kb][j] = xt[mrow][k];
                ah[kb][j] = hb[mrow][k];
            }
        }

        #pragma unroll
        for (int nti = 0; nti < 6; ++nti) {
            int nt = wv * 6 + nti;
            v8f accI = {}; v8f accH = {};
            #pragma unroll
            for (int kb = 0; kb < 4; ++kb) {
                accI = __builtin_amdgcn_wmma_f32_16x16x32_bf16(
                           false, ax[kb], false, wbi[nti][kb], (short)0, accI, false, false);
                accH = __builtin_amdgcn_wmma_f32_16x16x32_bf16(
                           false, ah[kb], false, wbh[nti][kb], (short)0, accH, false, false);
            }
            int n = nt * 16 + (lane & 15);
            #pragma unroll
            for (int v = 0; v < 8; ++v) {
                int m = v + 8 * hlf;
                si[m][n] = accI[v];
                sh[m][n] = accH[v];
            }
        }
        __syncthreads();

        #pragma unroll
        for (int m = 0; m < 16; ++m) {
            float r  = sigmoidf_((si[m][tid]      + bir) + (sh[m][tid]      + bhr));
            float u  = sigmoidf_((si[m][H_ + tid] + biu) + (sh[m][H_ + tid] + bhu)) * att_s[m];
            float ghn =           sh[m][2 * H_ + tid] + bhn;
            float ng = tanhf((si[m][2 * H_ + tid] + bin_) + r * ghn);
            float hprev = hreg[m];
            float hnew  = (1.0f - u) * hprev + u * ng;   // AUGRU: u gates NEW state
            bool valid  = t < lens[m];
            hreg[m] = valid ? hnew : hprev;
            hb[m][tid] = (__bf16)hreg[m];
        }
        __syncthreads();
    }

    #pragma unroll
    for (int m = 0; m < 16; ++m)
        out[(size_t)(b0 + m) * H_ + tid] = hreg[m];
}

// ---------------------------------------------------------------------------
// Attention MLP: per block = 32 (b,t) rows (2 M-tiles share each B-fragment).
// layer1 512->80 via WMMA (N padded to 96), layers 2/3 in VALU. Writes logits.
// ---------------------------------------------------------------------------
__global__ __launch_bounds__(64) void attn_mlp_kernel(
    const float* __restrict__ query, const __bf16* __restrict__ interests_bf,
    const __bf16* __restrict__ pW1, const float* __restrict__ b1,
    const float* __restrict__ W2, const float* __restrict__ b2,
    const float* __restrict__ Wf, const float* __restrict__ bfs,
    float* __restrict__ logits)
{
    __shared__ __bf16 din[32][512];
    __shared__ float  h1[32][80];
    __shared__ float  h2[32][40];

    const int tid  = threadIdx.x;
    const int lane = tid & 31;
    const int wv   = tid >> 5;            // 0..1
    const int row0 = blockIdx.x * 32;
    const int mrow = lane & 15;
    const int hlf  = lane >> 4;

    // build din = [q, k, q-k, q*k] (bf16) for 32 rows
    for (int idx = tid; idx < 32 * 512; idx += 64) {
        int m = idx >> 9, c = idx & 511;
        int part = c >> 7, k = c & 127;
        int row = row0 + m;
        int b = row / T_;
        float qv = query[(size_t)b * H_ + k];
        float kv = (float)interests_bf[(size_t)row * H_ + k];
        float val = (part == 0) ? qv
                  : (part == 1) ? kv
                  : (part == 2) ? (qv - kv) : (qv * kv);
        din[m][c] = (__bf16)val;
    }
    __syncthreads();

    // layer 1: two [16 x 96 x 512] WMMA M-tiles; wave0 -> nt 0..2, wave1 -> 3..5
    v8f acc0[3], acc1[3];
    #pragma unroll
    for (int i = 0; i < 3; ++i) { acc0[i] = (v8f){}; acc1[i] = (v8f){}; }
    for (int kb = 0; kb < 16; ++kb) {
        v16bf a0, a1;
        #pragma unroll
        for (int j = 0; j < 16; ++j) {
            int k = kb * 32 + a_klocal(j, hlf);
            a0[j] = din[mrow][k];
            a1[j] = din[16 + mrow][k];
        }
        #pragma unroll
        for (int i = 0; i < 3; ++i) {
            int nt = wv * 3 + i;
            const v16bf bw = *(const v16bf*)&pW1[(size_t)(((nt * 16 + kb) * 32) + lane) * 16];
            acc0[i] = __builtin_amdgcn_wmma_f32_16x16x32_bf16(
                          false, a0, false, bw, (short)0, acc0[i], false, false);
            acc1[i] = __builtin_amdgcn_wmma_f32_16x16x32_bf16(
                          false, a1, false, bw, (short)0, acc1[i], false, false);
        }
    }
    #pragma unroll
    for (int i = 0; i < 3; ++i) {
        int nt = wv * 3 + i;
        int n  = nt * 16 + (lane & 15);
        if (n < 80) {
            float bv = b1[n];
            #pragma unroll
            for (int v = 0; v < 8; ++v) {
                int m = v + 8 * hlf;
                h1[m][n]      = sigmoidf_(acc0[i][v] + bv);
                h1[16 + m][n] = sigmoidf_(acc1[i][v] + bv);
            }
        }
    }
    __syncthreads();

    // layer 2: 80 -> 40 (VALU)
    for (int idx = tid; idx < 32 * 40; idx += 64) {
        int m = idx / 40, o = idx % 40;
        float s = b2[o];
        #pragma unroll 4
        for (int i2 = 0; i2 < 80; ++i2) s += h1[m][i2] * W2[o * 80 + i2];
        h2[m][o] = sigmoidf_(s);
    }
    __syncthreads();

    // layer 3: 40 -> 1, scaled by 1/sqrt(H)
    if (tid < 32) {
        int m = tid;
        float s = bfs[0];
        #pragma unroll 4
        for (int i2 = 0; i2 < 40; ++i2) s += h2[m][i2] * Wf[i2];
        logits[row0 + m] = s * 0.08838834764831845f;  // 1/sqrt(128)
    }
}

// ---------------------------------------------------------------------------
// Masked softmax over T per batch row: one wave32 per row.
// ---------------------------------------------------------------------------
__global__ __launch_bounds__(32) void softmax_kernel(
    const float* __restrict__ logits, const int* __restrict__ keys_len,
    float* __restrict__ scores)
{
    int b = blockIdx.x, lane = threadIdx.x;
    int len = keys_len[b];
    float mx = -3.0e38f;
    for (int t = lane; t < T_; t += 32) {
        float l = (t < len) ? logits[(size_t)b * T_ + t] : -1.0e9f;
        mx = fmaxf(mx, l);
    }
    #pragma unroll
    for (int off = 16; off > 0; off >>= 1)
        mx = fmaxf(mx, __shfl_xor(mx, off, 32));
    float sum = 0.0f;
    for (int t = lane; t < T_; t += 32) {
        float l = (t < len) ? logits[(size_t)b * T_ + t] : -1.0e9f;
        sum += __expf(l - mx);
    }
    #pragma unroll
    for (int off = 16; off > 0; off >>= 1)
        sum += __shfl_xor(sum, off, 32);
    float inv = 1.0f / sum;
    for (int t = lane; t < T_; t += 32) {
        float l = (t < len) ? logits[(size_t)b * T_ + t] : -1.0e9f;
        scores[(size_t)b * T_ + t] = __expf(l - mx) * inv;
    }
}

// ---------------------------------------------------------------------------
extern "C" void kernel_launch(void* const* d_in, const int* in_sizes, int n_in,
                              void* d_out, int out_size, void* d_ws, size_t ws_size,
                              hipStream_t stream) {
    const float* query = (const float*)d_in[0];
    const float* keys  = (const float*)d_in[1];
    const int*   klen  = (const int*)d_in[2];
    const float* Wih_e = (const float*)d_in[3];
    const float* Whh_e = (const float*)d_in[4];
    const float* bih_e = (const float*)d_in[5];
    const float* bhh_e = (const float*)d_in[6];
    const float* Wih_a = (const float*)d_in[7];
    const float* Whh_a = (const float*)d_in[8];
    const float* bih_a = (const float*)d_in[9];
    const float* bhh_a = (const float*)d_in[10];
    const float* W1  = (const float*)d_in[11];
    const float* b1  = (const float*)d_in[12];
    const float* W2  = (const float*)d_in[13];
    const float* b2  = (const float*)d_in[14];
    const float* Wf  = (const float*)d_in[15];
    const float* bfs = (const float*)d_in[16];

    char* ws = (char*)d_ws;
    size_t off = 0;
    __bf16* interests_bf = (__bf16*)(ws + off); off += (size_t)B_ * T_ * H_ * sizeof(__bf16);
    float* logits    = (float*)(ws + off);  off += (size_t)B_ * T_ * sizeof(float);
    float* scores    = (float*)(ws + off);  off += (size_t)B_ * T_ * sizeof(float);
    __bf16* pWih_e   = (__bf16*)(ws + off); off += (size_t)GRU_PACK * sizeof(__bf16);
    __bf16* pWhh_e   = (__bf16*)(ws + off); off += (size_t)GRU_PACK * sizeof(__bf16);
    __bf16* pWih_a   = (__bf16*)(ws + off); off += (size_t)GRU_PACK * sizeof(__bf16);
    __bf16* pWhh_a   = (__bf16*)(ws + off); off += (size_t)GRU_PACK * sizeof(__bf16);
    __bf16* pW1      = (__bf16*)(ws + off); off += (size_t)W1_PACK  * sizeof(__bf16);

    int pack_total = 4 * GRU_PACK + W1_PACK;
    pack_weights_kernel<<<(pack_total + 255) / 256, 256, 0, stream>>>(
        Wih_e, Whh_e, Wih_a, Whh_a, W1, pWih_e, pWhh_e, pWih_a, pWhh_a, pW1);

    gru_scan_kernel<<<B_ / 16, 128, 0, stream>>>(
        keys, klen, bih_e, bhh_e, pWih_e, pWhh_e, interests_bf);

    attn_mlp_kernel<<<(B_ * T_) / 32, 64, 0, stream>>>(
        query, interests_bf, pW1, b1, W2, b2, Wf, bfs, logits);

    softmax_kernel<<<B_, 32, 0, stream>>>(logits, klen, scores);

    augru_scan_kernel<<<B_ / 16, 128, 0, stream>>>(
        interests_bf, scores, klen, bih_a, bhh_a, pWih_a, pWhh_a, (float*)d_out);
}
